// TurboQuantProd_42700564857506
// MI455X (gfx1250) — matrix operand order repaired and checked by reading, not compiled, not run on hardware
//
#include <hip/hip_runtime.h>
#include <math.h>

#define DIMK 128
#define LDX  132    // padded pitch for transposed/phase-2 operand buffer (bank-conflict-free)
#define TK   128
#define NQ   16
#define SKTOT 4096

typedef __bf16 bf16;
typedef __attribute__((ext_vector_type(16))) __bf16 v16bf;
typedef __attribute__((ext_vector_type(8)))  float  v8f;

// ---- WMMA fragment loaders (CDNA5 16x16x32 bf16 layouts, wave32) ----
// A-matrix 16x32 (MxK): lanes 0-15 -> M=lane, lanes 16-31 -> M=lane-16.
// v0..v3 hold K={0..7}(half0)/{8..15}(half1); v4..v7 hold K={16..23}/{24..31}.
__device__ __forceinline__ v16bf ld_a(const bf16* p, int m0, int k0, int ld, int lane) {
  const int half = lane >> 4;
  const int row  = m0 + (lane & 15);
  v16bf a;
#pragma unroll
  for (int e = 0; e < 16; ++e) {
    const int v  = e >> 1;
    const int kk = ((v < 4) ? 0 : 16) + half * 8 + ((v & 3) << 1) + (e & 1);
    a[e] = p[row * ld + k0 + kk];
  }
  return a;
}

__device__ __forceinline__ v16bf ld_a_gf32(const float* __restrict__ p, int m0, int k0,
                                           int ld, int lane) {
  const int half = lane >> 4;
  const int row  = m0 + (lane & 15);
  v16bf a;
#pragma unroll
  for (int e = 0; e < 16; ++e) {
    const int v  = e >> 1;
    const int kk = ((v < 4) ? 0 : 16) + half * 8 + ((v & 3) << 1) + (e & 1);
    a[e] = (bf16)p[row * ld + k0 + kk];
  }
  return a;
}

// B-matrix 32x16 (KxN): lane col N = lane&15; elems 0-7 hold K=8*half+e,
// elems 8-15 hold K=16+8*half+(e-8). Storage is K-contiguous: addr = (n0+N)*ld + k0+K,
// which the backend merges into ds_load_b128s.
__device__ __forceinline__ v16bf ld_b(const bf16* p, int k0, int n0, int ld, int lane) {
  const int half = lane >> 4;
  const int col  = n0 + (lane & 15);
  v16bf b;
#pragma unroll
  for (int e = 0; e < 16; ++e) {
    b[e] = p[col * ld + k0 + ((e < 8) ? 0 : 8) + half * 8 + e];
  }
  return b;
}

__device__ __forceinline__ v8f wmma_bf16(v16bf a, v16bf b, v8f c) {
  return __builtin_amdgcn_wmma_f32_16x16x32_bf16(false, a, false, b, (short)0, c,
                                                 false, false);
}

// 16(rows of A-strip) x 128 x 128 GEMM, software-pipelined via sched_group_barrier:
// issue the next kt's 16 ds_load_b128 while the current kt's 8 independent WMMAs run.
__device__ __forceinline__ void gemm_strip(const bf16* Abuf, int m0,
                                           const bf16* Bbuf, int ldb,
                                           int lane, v8f acc[8]) {
  v16bf a[4];
#pragma unroll
  for (int kt = 0; kt < 4; ++kt) a[kt] = ld_a(Abuf, m0, kt * 32, DIMK, lane);
#pragma unroll
  for (int kt = 0; kt < 4; ++kt) {
    v16bf b[8];
#pragma unroll
    for (int nt = 0; nt < 8; ++nt) b[nt] = ld_b(Bbuf, kt * 32, nt * 16, ldb, lane);
#pragma unroll
    for (int nt = 0; nt < 8; ++nt) acc[nt] = wmma_bf16(a[kt], b[nt], acc[nt]);
  }
  // Pinned schedule: 0x100 = DS read, 0x008 = MFMA/WMMA.
  // DS24(A+ktB0), DS16(ktB1), WMMA8(kt0), DS16(ktB2), WMMA8(kt1),
  // DS16(ktB3), WMMA8(kt2), WMMA8(kt3)  -> one-deep load/compute pipeline.
  __builtin_amdgcn_sched_group_barrier(0x100, 24, 0);
  __builtin_amdgcn_sched_group_barrier(0x100, 16, 0);
  __builtin_amdgcn_sched_group_barrier(0x008, 8, 0);
  __builtin_amdgcn_sched_group_barrier(0x100, 16, 0);
  __builtin_amdgcn_sched_group_barrier(0x008, 8, 0);
  __builtin_amdgcn_sched_group_barrier(0x100, 16, 0);
  __builtin_amdgcn_sched_group_barrier(0x008, 8, 0);
  __builtin_amdgcn_sched_group_barrier(0x008, 8, 0);
}

__global__ __launch_bounds__(256, 1)
void turboquant_score_kernel(const float* __restrict__ qry,
                             const float* __restrict__ keys,
                             const float* __restrict__ rot,
                             const float* __restrict__ smat,
                             const float* __restrict__ levels,
                             float* __restrict__ out) {
  // LDS budget (~135 KB of the 320 KB WGP pool):
  //   lds_R : R row-major   (B operand of GEMM1, value R[d][j] at d*128+j)   32 KB
  //   lds_X : R^T (padded)  (B operand of GEMM2), then S row-major (phase 2) 33 KB
  //   lds_A : Kn -> Yhat -> resid -> signs                                   32 KB
  //   lds_M : Kmse                                                           32 KB
  //   lds_Qp: Q @ S^T                                                         4 KB
  __shared__ bf16  lds_R [DIMK * DIMK];
  __shared__ bf16  lds_X [DIMK * LDX];
  __shared__ bf16  lds_A [TK * DIMK];
  __shared__ bf16  lds_M [TK * DIMK];
  __shared__ bf16  lds_Qp[NQ * DIMK];
  __shared__ float lds_norm[TK];
  __shared__ float lds_rn [TK];
  __shared__ float lds_red[256];

  const int tid  = threadIdx.x;
  const int lane = tid & 31;
  const int w    = tid >> 5;        // wave id 0..7
  const int kb   = blockIdx.x & 31; // key tile 0..31
  const int bh   = blockIdx.x >> 5; // fused (b,h) 0..63

  const float* __restrict__ Kg = keys + ((size_t)bh * SKTOT + (size_t)kb * TK) * DIMK;
  const float* __restrict__ Qg = qry + (size_t)bh * NQ * DIMK;
  float* __restrict__ Og       = out + (size_t)bh * NQ * SKTOT + (size_t)kb * TK;

  const float l0 = levels[0], l1 = levels[1], l2 = levels[2], l3 = levels[3];
  const float bd0 = 0.5f * (l0 + l1), bd1 = 0.5f * (l1 + l2), bd2 = 0.5f * (l2 + l3);

  // ---- stage 0: stage R and R^T (bf16), key norms (fp32), normalized keys ----
  for (int i = tid; i < DIMK * DIMK; i += 256) {
    const int r = i >> 7, c = i & 127;      // coalesced global read of rot[r][c]
    const bf16 v = (bf16)rot[i];
    lds_R[i] = v;                           // R row-major: addr r*128+c
    lds_X[c * LDX + r] = v;                 // R^T: value R[r][c] at c*132+r (padded)
  }
  {
    const int key   = tid >> 1;
    const int dbase = (tid & 1) * 64;
    const float4* p = (const float4*)(Kg + key * DIMK + dbase);
    float ss = 0.f;
#pragma unroll
    for (int i = 0; i < 16; ++i) {
      float4 v = p[i];
      ss += v.x * v.x + v.y * v.y + v.z * v.z + v.w * v.w;
    }
    lds_red[tid] = ss;
  }
  __syncthreads();
  if (tid < TK)
    lds_norm[tid] = fmaxf(sqrtf(lds_red[2 * tid] + lds_red[2 * tid + 1]), 1e-8f);
  __syncthreads();
  {
    const int key   = tid >> 1;
    const int dbase = (tid & 1) * 64;
    const float inv = 1.0f / lds_norm[key];
    const float4* p = (const float4*)(Kg + key * DIMK + dbase);
#pragma unroll
    for (int i = 0; i < 16; ++i) {
      float4 v = p[i];
      const int o   = key * DIMK + dbase + i * 4;
      lds_A[o + 0] = (bf16)(v.x * inv);
      lds_A[o + 1] = (bf16)(v.y * inv);
      lds_A[o + 2] = (bf16)(v.z * inv);
      lds_A[o + 3] = (bf16)(v.w * inv);
    }
  }
  __syncthreads();

  v8f acc[8];

  // ---- GEMM1: Y = Kn @ R^T  (wave w owns key rows [16w,16w+16)) ----
#pragma unroll
  for (int nt = 0; nt < 8; ++nt) acc[nt] = (v8f){};
  gemm_strip(lds_A, w * 16, lds_R, DIMK, lane, acc);   // B[j][d]=R[d][j]
  __syncthreads();
  // quantize to nearest Lloyd-Max level (searchsorted side='left' semantics)
#pragma unroll
  for (int nt = 0; nt < 8; ++nt) {
#pragma unroll
    for (int r = 0; r < 8; ++r) {
      const float y  = acc[nt][r];
      const float yh = (y <= bd0) ? l0 : (y <= bd1) ? l1 : (y <= bd2) ? l2 : l3;
      const int row = w * 16 + (lane >> 4) * 8 + r;
      const int col = nt * 16 + (lane & 15);
      lds_A[row * DIMK + col] = (bf16)yh;
    }
  }
  __syncthreads();

  // ---- GEMM2: Kmse = (Yhat @ R) * norm ; residual ----
#pragma unroll
  for (int nt = 0; nt < 8; ++nt) acc[nt] = (v8f){};
  gemm_strip(lds_A, w * 16, lds_X, LDX, lane, acc);    // B[j][d]=R[j][d] via R^T copy
  __syncthreads();
#pragma unroll
  for (int nt = 0; nt < 8; ++nt) {
#pragma unroll
    for (int r = 0; r < 8; ++r) {
      const int row = w * 16 + (lane >> 4) * 8 + r;  // key
      const int col = nt * 16 + (lane & 15);         // d
      const float km  = acc[nt][r] * lds_norm[row];
      const float res = Kg[row * DIMK + col] - km;   // fp32 residual (L2-hot reload)
      lds_M[row * DIMK + col] = (bf16)km;
      lds_A[row * DIMK + col] = (bf16)res;
    }
  }
  __syncthreads();

  // ---- phase 2: load S row-major into lds_X; rnorm from residual rows ----
  for (int i = tid; i < DIMK * DIMK; i += 256) {
    const int r = i >> 7, c = i & 127;
    lds_X[r * LDX + c] = (bf16)smat[i];     // S[m][j] at m*132+j (S^T is K-contiguous)
  }
  if (tid < TK) {
    const bf16* rp = lds_A + tid * DIMK;
    float ss = 0.f;
#pragma unroll
    for (int j = 0; j < DIMK; ++j) {
      const float v = (float)rp[j];
      ss += v * v;
    }
    lds_rn[tid] = fmaxf(sqrtf(ss), 1e-10f);
  }
  __syncthreads();

  // ---- GEMM4 + Qp: share the Q A-fragments (loaded from global once) ----
  v8f cmse = {};
  {
    v16bf aq[4];
#pragma unroll
    for (int kt = 0; kt < 4; ++kt) aq[kt] = ld_a_gf32(Qg, 0, kt * 32, DIMK, lane);

    // scores_mse tile (16q x 16k, N-tile = w):  B[d][key]=Kmse[key][d]
    v8f cq = {};
#pragma unroll
    for (int kt = 0; kt < 4; ++kt) {
      v16bf bm = ld_b(lds_M, kt * 32, w * 16, DIMK, lane);
      cmse = wmma_bf16(aq[kt], bm, cmse);
      v16bf bs = ld_b(lds_X, kt * 32, w * 16, LDX, lane);   // B[j][m]=S[m][j]
      cq = wmma_bf16(aq[kt], bs, cq);
    }
    // Qp = Q @ S^T, N-tile w -> stash bf16 in LDS for GEMM5
#pragma unroll
    for (int r = 0; r < 8; ++r) {
      const int row = (lane >> 4) * 8 + r;   // q
      const int col = w * 16 + (lane & 15);  // m
      lds_Qp[row * DIMK + col] = (bf16)cq[r];
    }
  }

  // ---- GEMM3: proj = resid @ S^T  ->  signs (sign invariant to 1/rnorm) ----
#pragma unroll
  for (int nt = 0; nt < 8; ++nt) acc[nt] = (v8f){};
  gemm_strip(lds_A, w * 16, lds_X, LDX, lane, acc);    // B[j][m]=S[m][j]
  __syncthreads();
#pragma unroll
  for (int nt = 0; nt < 8; ++nt) {
#pragma unroll
    for (int r = 0; r < 8; ++r) {
      const int row = w * 16 + (lane >> 4) * 8 + r;
      const int col = nt * 16 + (lane & 15);
      lds_A[row * DIMK + col] = (bf16)((acc[nt][r] >= 0.f) ? 1.0f : -1.0f);
    }
  }
  __syncthreads();

  // ---- GEMM5: qjl = Qp @ signs^T ; fuse and store ----
  {
    v8f cq = {};
#pragma unroll
    for (int kt = 0; kt < 4; ++kt) {
      v16bf aq = ld_a(lds_Qp, 0, kt * 32, DIMK, lane);
      v16bf bs = ld_b(lds_A, kt * 32, w * 16, DIMK, lane);  // B[m][key]=signs[key][m]
      cq = wmma_bf16(aq, bs, cq);
    }
    const float scale = 0.009791516697777346f;  // sqrt(pi/2)/128
#pragma unroll
    for (int r = 0; r < 8; ++r) {
      const int qrow = (lane >> 4) * 8 + r;
      const int keyl = w * 16 + (lane & 15);
      Og[(size_t)qrow * SKTOT + keyl] = cmse[r] + scale * lds_rn[keyl] * cq[r];
    }
  }
}

extern "C" void kernel_launch(void* const* d_in, const int* in_sizes, int n_in,
                              void* d_out, int out_size, void* d_ws, size_t ws_size,
                              hipStream_t stream) {
  (void)in_sizes; (void)n_in; (void)d_ws; (void)ws_size; (void)out_size;
  const float* qry    = (const float*)d_in[0];
  const float* keys   = (const float*)d_in[1];
  const float* rot    = (const float*)d_in[2];
  const float* smat   = (const float*)d_in[3];
  const float* levels = (const float*)d_in[4];
  float* out = (float*)d_out;

  // grid = (B*H) * (SK / TK) = 64 * 32
  dim3 grid(2048), block(256);
  hipLaunchKernelGGL(turboquant_score_kernel, grid, block, 0, stream,
                     qry, keys, rot, smat, levels, out);
}
